// MoECompositionalFFN_6167573037669
// MI455X (gfx1250) — matrix operand order repaired and checked by reading, compiled, not run on hardware
//
#include <hip/hip_runtime.h>
#include <math.h>
#include <stdint.h>

// ---------------------------------------------------------------------------
// Problem constants (from reference)
// ---------------------------------------------------------------------------
#define T_TOKENS 16384   // BATCH * SEQ
#define DMODEL   512
#define DFF      2048
#define N_EXP    8
#define N_PRIM   16
#define RANKR    64
#define KRANK    256     // TOP_K_PRIM * RANK

// ---------------------------------------------------------------------------
// WMMA types (CDNA5 / gfx1250, wave32)
// ---------------------------------------------------------------------------
typedef __attribute__((ext_vector_type(16))) __bf16 v16bf;
typedef __attribute__((ext_vector_type(8)))  float  v8f;

union Frag16 { v16bf v; uint4 q[2]; };

// B-matrix 32x16 bf16 fragment (row-major B KxN, leading dim ldb):
//   lane L holds column N = L%16, K 8h..8h+7 then 16+8h..16+8h+7 (h = L/16).
__device__ __forceinline__ v16bf load_b_frag(const __bf16* B, int ldb, int k0,
                                             int n0, int lane) {
  const __bf16* p = B + (size_t)(k0 + ((lane >> 4) << 3)) * ldb + n0 + (lane & 15);
  v16bf v;
  #pragma unroll
  for (int s = 0; s < 8; ++s) {
    v[s]     = p[(size_t)s * ldb];
    v[s + 8] = p[(size_t)(s + 16) * ldb];
  }
  return v;
}

__device__ __forceinline__ float gelu_exact(float x) {
  return 0.5f * x * (1.0f + erff(x * 0.70710678118654752440f));
}

// Raw workgroup split-barrier (no implicit LOADcnt/STOREcnt drain, unlike the
// fence implied by __syncthreads). Cross-wave LDS visibility is provided by
// each wave's own s_wait_asynccnt / s_wait_dscnt before signaling.
__device__ __forceinline__ void wg_barrier() {
  asm volatile("s_barrier_signal -1\n\ts_barrier_wait -1" ::: "memory");
}

enum { EPI_STORE = 0, EPI_GELU = 1, EPI_GATE = 2 };

// ---------------------------------------------------------------------------
// Generic WMMA GEMM: C[M x N] = A[M x K] * B[K x N], bf16 in / f32 accum.
// Block = 256 threads = 8 waves; block tile = 32(M) x 256(N); wave = 16 x 64.
// The A k-slice (32x32 bf16, 2KB) is staged in LDS once per block via
// GLOBAL_LOAD_ASYNC_TO_LDS (ASYNCcnt) and shared by all 8 waves; LDS A-tiles
// and B register fragments are double-buffered so DMA + global loads overlap
// the WMMAs. Grid: (N/256, M/32); all dims divide evenly -> EXEC all ones.
// ---------------------------------------------------------------------------
template<int EPI>
__global__ __launch_bounds__(256) void gemm_bf16_wmma(
    const __bf16* __restrict__ A, const __bf16* __restrict__ B,
    __bf16* __restrict__ Cb, float* __restrict__ Cf,
    const float* __restrict__ gate, int expert, int overwrite,
    int K, int lda, int ldb, int ldc)
{
  __shared__ alignas(16) __bf16 smem[2][32 * 32];   // 2 x 2KB A tiles

  const int lane = threadIdx.x & 31;
  const int wave = threadIdx.x >> 5;
  const int wm   = (wave >> 2) << 4;                // 0 or 16 (M sub-tile)
  const int m0   = blockIdx.y * 32;
  const int n0   = blockIdx.x * 256 + (wave & 3) * 64;

  // Async DMA one 32x32 bf16 A tile into LDS buffer bufi (b64 per lane).
  auto issue_a = [&](int k0, int bufi) {
    const int l   = threadIdx.x;                    // 0..255
    const int row = l >> 3;                         // 0..31
    const int c4  = (l & 7) << 2;                   // 0,4,...,28 (bf16 elems)
    const __bf16* g = A + (size_t)(m0 + row) * lda + (k0 + c4);
    // LDS aperture: low 32 bits of the flat address are the LDS byte offset.
    const unsigned loff =
        (unsigned)(uintptr_t)(const void*)&smem[bufi][row * 32 + c4];
    asm volatile("global_load_async_to_lds_b64 %0, %1, off"
                 :: "v"(loff), "v"(g) : "memory");
  };

  // A-fragment from the LDS tile, ISA 16-bit A layout:
  //   lanes 0-15 : M=lane,    K kh..kh+7 then kh+16..kh+23 (kh=0)
  //   lanes 16-31: M=lane-16, same with kh=8
  auto lds_a = [&](int bufi) -> v16bf {
    const __bf16* p = &smem[bufi][(wm + (lane & 15)) * 32 + ((lane >> 4) << 3)];
    Frag16 f;
    f.q[0] = *reinterpret_cast<const uint4*>(p);
    f.q[1] = *reinterpret_cast<const uint4*>(p + 16);
    return f.v;
  };

  v8f acc[4];
  #pragma unroll
  for (int t = 0; t < 4; ++t)
    #pragma unroll
    for (int r = 0; r < 8; ++r) acc[t][r] = 0.0f;

  const int niter = K / 32;                         // 8 / 16 / 64: always even
  v16bf b0[4], b1[4];

  // Prologue: DMA A tile 0, load B fragments for k=0.
  issue_a(0, 0);
  #pragma unroll
  for (int t = 0; t < 4; ++t) b0[t] = load_b_frag(B, ldb, 0, n0 + t * 16, lane);

  auto kstep = [&](int i, v16bf (&bc)[4], v16bf (&bn)[4]) {
    const int k0 = i * 32;
    const bool haveNext = (i + 1 < niter);          // uniform branch
    if (haveNext) {
      issue_a(k0 + 32, (i + 1) & 1);                // safe: barrier below
      #pragma unroll
      for (int t = 0; t < 4; ++t)
        bn[t] = load_b_frag(B, ldb, k0 + 32, n0 + t * 16, lane);
      if (k0 + 64 < K)
        __builtin_prefetch((const void*)(B + (size_t)(k0 + 64 + lane) * ldb + n0), 0, 1);
      asm volatile("s_wait_asynccnt 0x1" ::: "memory");  // tile i landed
    } else {
      asm volatile("s_wait_asynccnt 0x0" ::: "memory");
    }
    wg_barrier();                 // all waves' DMA for tile i visible
    const v16bf a = lds_a(i & 1);
    #pragma unroll
    for (int t = 0; t < 4; ++t)
      acc[t] = __builtin_amdgcn_wmma_f32_16x16x32_bf16(
          false, a, false, bc[t], (short)0, acc[t], false, false);
    // Retire this wave's LDS reads of buf[i&1] BEFORE signaling buffer reuse
    // (WMMAs are register-only and may be scheduled past the barrier asm).
    asm volatile("s_wait_dscnt 0x0" ::: "memory");
    wg_barrier();                 // tile i buffer free for the next DMA
  };

  for (int i = 0; i < niter; i += 2) {
    kstep(i,     b0, b1);
    kstep(i + 1, b1, b0);
  }

  // C/D layout: VGPR r, lanes 0-15 -> M=r, N=lane; lanes 16-31 -> M=8+r.
  const int mrow = m0 + wm + ((lane >> 4) << 3);
  const int ncol = lane & 15;
  #pragma unroll
  for (int t = 0; t < 4; ++t) {
    const int n = n0 + t * 16 + ncol;
    #pragma unroll
    for (int r = 0; r < 8; ++r) {
      float v = acc[t][r];
      const int m = mrow + r;
      if (EPI == EPI_GELU) {
        v = gelu_exact(v);
        Cb[(size_t)m * ldc + n] = (__bf16)v;
      } else if (EPI == EPI_GATE) {
        const float g = gate[(size_t)m * N_EXP + expert];
        float* o = Cf + (size_t)m * ldc + n;
        if (overwrite) *o = g * v; else *o += g * v;
      } else {
        Cb[(size_t)m * ldc + n] = (__bf16)v;
      }
    }
  }
}

// ---------------------------------------------------------------------------
// Primitive composition: per-expert softmax over 16 logits, top-4, renorm,
// sqrt weights. Also zeroes the aux-loss accumulators.
// ---------------------------------------------------------------------------
__global__ void compose_topk_kernel(const float* __restrict__ fc1,
                                    const float* __restrict__ fc2,
                                    int* __restrict__ idx1, float* __restrict__ w1,
                                    int* __restrict__ idx2, float* __restrict__ w2,
                                    float* __restrict__ psum, float* __restrict__ cnt)
{
  const int tid = threadIdx.x;
  if (tid < N_EXP) { psum[tid] = 0.0f; cnt[tid] = 0.0f; }
  if (tid >= N_EXP) return;

  const float* srcs[2] = { fc1, fc2 };
  int*   idxs[2] = { idx1, idx2 };
  float* wgts[2] = { w1, w2 };

  for (int s = 0; s < 2; ++s) {
    const float* lg = srcs[s] + tid * N_PRIM;
    float p[N_PRIM];
    float mx = lg[0];
    for (int i = 1; i < N_PRIM; ++i) mx = fmaxf(mx, lg[i]);
    float se = 0.0f;
    for (int i = 0; i < N_PRIM; ++i) { p[i] = expf(lg[i] - mx); se += p[i]; }
    const float inv = 1.0f / se;
    for (int i = 0; i < N_PRIM; ++i) p[i] *= inv;

    int  ti[4]; float tw[4]; bool used[N_PRIM];
    for (int i = 0; i < N_PRIM; ++i) used[i] = false;
    for (int kk = 0; kk < 4; ++kk) {
      int best = -1; float bv = -1.0f;
      for (int i = 0; i < N_PRIM; ++i)
        if (!used[i] && p[i] > bv) { bv = p[i]; best = i; }
      used[best] = true; ti[kk] = best; tw[kk] = bv;
    }
    const float ssum = tw[0] + tw[1] + tw[2] + tw[3] + 1e-8f;
    for (int kk = 0; kk < 4; ++kk) {
      idxs[s][tid * 4 + kk] = ti[kk];
      wgts[s][tid * 4 + kk] = sqrtf(tw[kk] / ssum + 1e-8f);
    }
  }
}

// A_cat[e, d, kk*64+r] = A[idx[e,kk], d, r] * w[e,kk]  -> bf16
__global__ void build_Acat_kernel(const float* __restrict__ Ain,
                                  const int* __restrict__ idx,
                                  const float* __restrict__ w,
                                  __bf16* __restrict__ out, int din)
{
  const long long total = (long long)N_EXP * din * KRANK;
  long long i = (long long)blockIdx.x * blockDim.x + threadIdx.x;
  const long long stride = (long long)gridDim.x * blockDim.x;
  for (; i < total; i += stride) {
    const int j = (int)(i % KRANK);
    const long long rem = i / KRANK;
    const int d = (int)(rem % din);
    const int e = (int)(rem / din);
    const int kk = j >> 6, r = j & 63;
    const float v = Ain[((size_t)idx[e * 4 + kk] * din + d) * RANKR + r] * w[e * 4 + kk];
    out[i] = (__bf16)v;
  }
}

// B_cat[e, kk*64+r, c] = B[idx[e,kk], r, c] * w[e,kk]  -> bf16
__global__ void build_Bcat_kernel(const float* __restrict__ Bin,
                                  const int* __restrict__ idx,
                                  const float* __restrict__ w,
                                  __bf16* __restrict__ out, int dout)
{
  const long long total = (long long)N_EXP * KRANK * dout;
  long long i = (long long)blockIdx.x * blockDim.x + threadIdx.x;
  const long long stride = (long long)gridDim.x * blockDim.x;
  for (; i < total; i += stride) {
    const int c = (int)(i % dout);
    const long long rem = i / dout;
    const int j = (int)(rem % KRANK);
    const int e = (int)(rem / KRANK);
    const int kk = j >> 6, r = j & 63;
    const float v = Bin[((size_t)idx[e * 4 + kk] * RANKR + r) * dout + c] * w[e * 4 + kk];
    out[i] = (__bf16)v;
  }
}

__global__ void f32_to_bf16_kernel(const float* __restrict__ in,
                                   __bf16* __restrict__ out, long long n)
{
  long long i = (long long)blockIdx.x * blockDim.x + threadIdx.x;
  const long long stride = (long long)gridDim.x * blockDim.x;
  for (; i < n; i += stride) out[i] = (__bf16)in[i];
}

// ---------------------------------------------------------------------------
// Router: logits = x @ Wr^T, softmax over 8, top-2 renormalized gate,
// plus LDS-reduced aux-loss partials (prob sums, token counts).
// ---------------------------------------------------------------------------
__global__ __launch_bounds__(256) void router_kernel(
    const float* __restrict__ x, const float* __restrict__ Wr,
    float* __restrict__ gate, float* __restrict__ psum, float* __restrict__ cnt)
{
  __shared__ float s_psum[N_EXP];
  __shared__ float s_cnt[N_EXP];
  if (threadIdx.x < N_EXP) { s_psum[threadIdx.x] = 0.0f; s_cnt[threadIdx.x] = 0.0f; }
  __syncthreads();

  const int t = blockIdx.x * blockDim.x + threadIdx.x;   // grid exactly covers T
  const float* xr = x + (size_t)t * DMODEL;

  float logits[N_EXP];
  #pragma unroll
  for (int e = 0; e < N_EXP; ++e) {
    const float* wr = Wr + e * DMODEL;
    float s = 0.0f;
    for (int d = 0; d < DMODEL; d += 4) {
      const float4 xv = *reinterpret_cast<const float4*>(xr + d);
      const float4 wv = *reinterpret_cast<const float4*>(wr + d);
      s += xv.x * wv.x + xv.y * wv.y + xv.z * wv.z + xv.w * wv.w;
    }
    logits[e] = s;
  }

  float mx = logits[0];
  #pragma unroll
  for (int e = 1; e < N_EXP; ++e) mx = fmaxf(mx, logits[e]);
  float p[N_EXP]; float se = 0.0f;
  #pragma unroll
  for (int e = 0; e < N_EXP; ++e) { p[e] = expf(logits[e] - mx); se += p[e]; }
  const float inv = 1.0f / se;
  #pragma unroll
  for (int e = 0; e < N_EXP; ++e) p[e] *= inv;

  int i1 = 0;
  #pragma unroll
  for (int e = 1; e < N_EXP; ++e) if (p[e] > p[i1]) i1 = e;
  int i2 = (i1 == 0) ? 1 : 0;
  #pragma unroll
  for (int e = 0; e < N_EXP; ++e) if (e != i1 && p[e] > p[i2]) i2 = e;

  float w1 = p[i1], w2 = p[i2];
  const float s2 = w1 + w2 + 1e-8f;
  w1 /= s2; w2 /= s2;

  #pragma unroll
  for (int e = 0; e < N_EXP; ++e) {
    const float gv = (e == i1) ? w1 : ((e == i2) ? w2 : 0.0f);
    gate[(size_t)t * N_EXP + e] = gv;
    atomicAdd(&s_psum[e], p[e]);
  }
  atomicAdd(&s_cnt[i1], 1.0f);
  atomicAdd(&s_cnt[i2], 1.0f);
  __syncthreads();
  if (threadIdx.x < N_EXP) {
    atomicAdd(&psum[threadIdx.x], s_psum[threadIdx.x]);
    atomicAdd(&cnt[threadIdx.x],  s_cnt[threadIdx.x]);
  }
}

__global__ void aux_finalize_kernel(const float* __restrict__ psum,
                                    const float* __restrict__ cnt,
                                    float* __restrict__ aux)
{
  if (threadIdx.x == 0 && blockIdx.x == 0) {
    float tot = 0.0f;
    for (int e = 0; e < N_EXP; ++e) tot += cnt[e];
    float a = 0.0f;
    for (int e = 0; e < N_EXP; ++e) {
      const float f = cnt[e] / (tot + 1e-8f);
      const float P = psum[e] / (float)T_TOKENS;
      a += f * P;
    }
    aux[0] = (float)N_EXP * a;
  }
}

// ---------------------------------------------------------------------------
// Host-side launch
// ---------------------------------------------------------------------------
extern "C" void kernel_launch(void* const* d_in, const int* in_sizes, int n_in,
                              void* d_out, int out_size, void* d_ws, size_t ws_size,
                              hipStream_t stream) {
  (void)in_sizes; (void)n_in; (void)out_size; (void)ws_size;

  const float* x   = (const float*)d_in[0];   // (T, 512)
  const float* Wr  = (const float*)d_in[1];   // (8, 512)
  const float* fc1 = (const float*)d_in[2];   // (8, 16)
  const float* fc2 = (const float*)d_in[3];   // (8, 16)
  const float* A1  = (const float*)d_in[4];   // (16, 512, 64)
  const float* B1  = (const float*)d_in[5];   // (16, 64, 2048)
  const float* A2  = (const float*)d_in[6];   // (16, 2048, 64)
  const float* B2  = (const float*)d_in[7];   // (16, 64, 512)

  float* out = (float*)d_out;                       // (T, 512)
  float* aux = out + (size_t)T_TOKENS * DMODEL;     // scalar

  // Bump allocator over d_ws (256B aligned chunks). Total ~114 MB.
  char* base = (char*)d_ws;
  size_t off = 0;
  auto alloc = [&](size_t bytes) -> void* {
    off = (off + 255) & ~(size_t)255;
    void* p = base + off;
    off += bytes;
    return p;
  };

  float* gate = (float*)alloc((size_t)T_TOKENS * N_EXP * sizeof(float));
  float* psum = (float*)alloc(N_EXP * sizeof(float));
  float* cnt  = (float*)alloc(N_EXP * sizeof(float));
  int*   idx1 = (int*)  alloc(N_EXP * 4 * sizeof(int));
  float* w1   = (float*)alloc(N_EXP * 4 * sizeof(float));
  int*   idx2 = (int*)  alloc(N_EXP * 4 * sizeof(int));
  float* w2   = (float*)alloc(N_EXP * 4 * sizeof(float));
  __bf16* x_bf = (__bf16*)alloc((size_t)T_TOKENS * DMODEL * 2);
  __bf16* A1c  = (__bf16*)alloc((size_t)N_EXP * DMODEL * KRANK * 2);
  __bf16* B1c  = (__bf16*)alloc((size_t)N_EXP * KRANK * DFF * 2);
  __bf16* A2c  = (__bf16*)alloc((size_t)N_EXP * DFF * KRANK * 2);
  __bf16* B2c  = (__bf16*)alloc((size_t)N_EXP * KRANK * DMODEL * 2);
  __bf16* t1   = (__bf16*)alloc((size_t)T_TOKENS * KRANK * 2);   // reused as t2
  __bf16* h    = (__bf16*)alloc((size_t)T_TOKENS * DFF * 2);
  __bf16* t2 = t1;  // t1 is dead once h is produced

  // 1) composition weights (also zeroes psum/cnt before router)
  compose_topk_kernel<<<1, 32, 0, stream>>>(fc1, fc2, idx1, w1, idx2, w2, psum, cnt);

  // 2) composed bf16 weight banks
  build_Acat_kernel<<<2048, 256, 0, stream>>>(A1, idx1, w1, A1c, DMODEL);
  build_Bcat_kernel<<<4096, 256, 0, stream>>>(B1, idx1, w1, B1c, DFF);
  build_Acat_kernel<<<4096, 256, 0, stream>>>(A2, idx2, w2, A2c, DFF);
  build_Bcat_kernel<<<2048, 256, 0, stream>>>(B2, idx2, w2, B2c, DMODEL);

  // 3) x -> bf16
  f32_to_bf16_kernel<<<8192, 256, 0, stream>>>(x, x_bf, (long long)T_TOKENS * DMODEL);

  // 4) router + aux loss
  router_kernel<<<T_TOKENS / 256, 256, 0, stream>>>(x, Wr, gate, psum, cnt);
  aux_finalize_kernel<<<1, 32, 0, stream>>>(psum, cnt, aux);

  // 5) expert loop: four chained WMMA GEMMs per expert
  const dim3 blk(256);
  for (int e = 0; e < N_EXP; ++e) {
    // t1 = x_bf @ A1c[e]          (T x 256, K=512)
    gemm_bf16_wmma<EPI_STORE><<<dim3(1, T_TOKENS / 32), blk, 0, stream>>>(
        x_bf, A1c + (size_t)e * DMODEL * KRANK, t1, nullptr, nullptr, 0, 0,
        DMODEL, DMODEL, KRANK, KRANK);
    // h = gelu(t1 @ B1c[e])       (T x 2048, K=256)
    gemm_bf16_wmma<EPI_GELU><<<dim3(DFF / 256, T_TOKENS / 32), blk, 0, stream>>>(
        t1, B1c + (size_t)e * KRANK * DFF, h, nullptr, nullptr, 0, 0,
        KRANK, KRANK, DFF, DFF);
    // t2 = h @ A2c[e]             (T x 256, K=2048)
    gemm_bf16_wmma<EPI_STORE><<<dim3(1, T_TOKENS / 32), blk, 0, stream>>>(
        h, A2c + (size_t)e * DFF * KRANK, t2, nullptr, nullptr, 0, 0,
        DFF, DFF, KRANK, KRANK);
    // out (=/+)= gate[:,e] * (t2 @ B2c[e])   (T x 512, K=256)
    gemm_bf16_wmma<EPI_GATE><<<dim3(DMODEL / 256, T_TOKENS / 32), blk, 0, stream>>>(
        t2, B2c + (size_t)e * KRANK * DMODEL, nullptr, out, gate, e, (e == 0) ? 1 : 0,
        KRANK, KRANK, DMODEL, DMODEL);
  }
}
